// SDOT_16466904613557
// MI455X (gfx1250) — matrix operand order repaired and checked by reading, not compiled, not run on hardware
//
#include <hip/hip_runtime.h>

// ---------------------------------------------------------------------------
// Entropic semi-dual OT objective for MI455X (gfx1250, wave32, WMMA).
//   out[n] = -eps*logsumexp_m((psi[m]-||x_n-y_m||^2)/eps) + mean(psi)
// Main GEMM (8192x16384x128) runs on v_wmma_f32_16x16x32_bf16 using a
// split-2 bf16 emulation of f32 (hi*hi + lo*hi + hi*lo), fused with a
// streaming base-2 online logsumexp so the 8Kx16K score matrix never
// touches memory. 8 waves share each staged LDS chunk; two accumulator
// chains per inner loop give the XDL scheduler independent work.
// ---------------------------------------------------------------------------

typedef __attribute__((ext_vector_type(16))) __bf16 v16bf;
typedef __attribute__((ext_vector_type(8)))  __bf16 v8bf;
typedef __attribute__((ext_vector_type(8)))  float  v8f;

namespace {
constexpr int   kN      = 8192;
constexpr int   kM      = 16384;
constexpr int   kK      = 128;
constexpr int   kMS     = 8;              // M-split across grid.y
constexpr int   kRowsWG = 128;            // rows of N per workgroup (8 waves x 16)
constexpr int   kMChunk = 64;             // target rows staged in LDS per step
constexpr int   kTPB    = 256;            // 8 waves (wave32)
constexpr int   kMRange = kM / kMS;       // 2048 columns per workgroup
constexpr int   kNChunk = kMRange / kMChunk;   // 32 chunks
constexpr int   kPitch  = kK + 8;         // bf16 pitch -> 272B row, bank-shift 4
constexpr float kInvEln2 = 28.853900817779268f;     // 1/(eps*ln2), eps=0.05
constexpr float kScale2  = 2.0f * kInvEln2;         // 2/(eps*ln2)
constexpr float kNegEln2 = -0.034657359027997264f;  // -eps*ln2
}

__device__ __forceinline__ void bf16_split(float f, __bf16& h, __bf16& l) {
    h = (__bf16)f;
    l = (__bf16)(f - (float)h);
}

// ---- prep: x_sq[n] = ||x_n||^2 ---------------------------------------------
__global__ __launch_bounds__(256) void ot_prep_xsq(const float* __restrict__ src,
                                                   float* __restrict__ xsq) {
    int n = blockIdx.x * blockDim.x + threadIdx.x;
    const float4* r = (const float4*)(src + (size_t)n * kK);
    float s = 0.f;
#pragma unroll
    for (int i = 0; i < kK / 4; ++i) {
        float4 v = r[i];
        s += v.x * v.x + v.y * v.y + v.z * v.z + v.w * v.w;
    }
    xsq[n] = s;
}

// ---- prep: tcoef[m] = (psi[m] - ||y_m||^2) / (eps*ln2) ---------------------
__global__ __launch_bounds__(256) void ot_prep_tcoef(const float* __restrict__ tgt,
                                                     const float* __restrict__ psi,
                                                     float* __restrict__ tco) {
    int m = blockIdx.x * blockDim.x + threadIdx.x;
    const float4* r = (const float4*)(tgt + (size_t)m * kK);
    float s = 0.f;
#pragma unroll
    for (int i = 0; i < kK / 4; ++i) {
        float4 v = r[i];
        s += v.x * v.x + v.y * v.y + v.z * v.z + v.w * v.w;
    }
    tco[m] = (psi[m] - s) * kInvEln2;
}

// ---- prep: mean(psi) -------------------------------------------------------
__global__ __launch_bounds__(256) void ot_prep_mean(const float* __restrict__ psi,
                                                    float* __restrict__ meanp) {
    __shared__ float red[256];
    float s = 0.f;
    for (int i = threadIdx.x; i < kM; i += 256) s += psi[i];
    red[threadIdx.x] = s;
    __syncthreads();
    for (int o = 128; o > 0; o >>= 1) {
        if (threadIdx.x < (unsigned)o) red[threadIdx.x] += red[threadIdx.x + o];
        __syncthreads();
    }
    if (threadIdx.x == 0) meanp[0] = red[0] / (float)kM;
}

// ---- main fused WMMA-GEMM + streaming logsumexp ----------------------------
__global__ __launch_bounds__(kTPB) void ot_main(const float* __restrict__ src,
                                                const float* __restrict__ tgt,
                                                const float* __restrict__ tco,
                                                float* __restrict__ pmax,
                                                float* __restrict__ psum) {
    __shared__ __bf16 sBh[kMChunk][kPitch];
    __shared__ __bf16 sBl[kMChunk][kPitch];
    __shared__ float  sT[kMChunk];

    const int tid  = threadIdx.x;
    const int lane = tid & 31;
    const int wv   = tid >> 5;
    const int lh   = lane >> 4;     // 0 = lanes 0-15, 1 = lanes 16-31
    const int l15  = lane & 15;
    const int rowbase = blockIdx.x * kRowsWG + wv * 16;
    const int row     = rowbase + l15;
    const int mbase0  = blockIdx.y * kMRange;

    // --- A slab (16 rows x 128 K) -> registers, bf16 hi/lo, WMMA A-layout ---
    // A 16x32, contiguous k order chosen here (B matches): lane holds row (lane%16); element j -> K = base + hi8 +
    // j, hi8 = 16 for lanes 16-31.
    v16bf ah[4], al[4];
    {
        const float* srow = src + (size_t)row * kK;
        const int hi8 = lh * 16;
#pragma unroll
        for (int kc = 0; kc < 4; ++kc) {
            const int base = kc * 32 + hi8;
            float4 f0 = *(const float4*)(srow + base);
            float4 f1 = *(const float4*)(srow + base + 4);
            float4 f2 = *(const float4*)(srow + base + 8);
            float4 f3 = *(const float4*)(srow + base + 12);
            float t[16] = { f0.x, f0.y, f0.z, f0.w, f1.x, f1.y, f1.z, f1.w,
                            f2.x, f2.y, f2.z, f2.w, f3.x, f3.y, f3.z, f3.w };
#pragma unroll
            for (int j = 0; j < 16; ++j) {
                __bf16 h, l;
                bf16_split(t[j], h, l);
                ah[kc][j] = h;
                al[kc][j] = l;
            }
        }
    }

    // online logsumexp state: slot i <-> local rows {i, i+8} (C VGPR layout)
    float rmax[8], rsum[8];
#pragma unroll
    for (int i = 0; i < 8; ++i) { rmax[i] = -3.0e38f; rsum[i] = 0.f; }

    for (int ch = 0; ch < kNChunk; ++ch) {
        const int mbase = mbase0 + ch * kMChunk;
        __syncthreads();
        // cooperative load + split of 64x128 f32 target chunk into LDS bf16;
        // 8 elements per step -> 16B ds stores.
#pragma unroll
        for (int t = 0; t < 4; ++t) {
            int idx = tid + t * kTPB;           // 0..1023 8-float segments
            int r   = idx >> 4;                 // 16 segments per row
            int sg  = idx & 15;
            const float* gp = tgt + (size_t)(mbase + r) * kK + sg * 8;
            float4 g0 = *(const float4*)gp;
            float4 g1 = *(const float4*)(gp + 4);
            union { __bf16 b[8]; uint4 u; } H, L;
            float e[8] = { g0.x, g0.y, g0.z, g0.w, g1.x, g1.y, g1.z, g1.w };
#pragma unroll
            for (int j = 0; j < 8; ++j) bf16_split(e[j], H.b[j], L.b[j]);
            *(uint4*)&sBh[r][sg * 8] = H.u;
            *(uint4*)&sBl[r][sg * 8] = L.u;
        }
        if (tid < kMChunk) sT[tid] = tco[mbase + tid];
        if (ch + 1 < kNChunk && tid < 2 * kMChunk) {  // prefetch next chunk
            const float* nf = tgt + (size_t)(mbase + kMChunk + (tid >> 1)) * kK
                              + (tid & 1) * 64;
            __builtin_prefetch(nf, 0, 0);
        }
        __syncthreads();

        // two 16-column tiles at a time -> two independent WMMA chains
#pragma unroll
        for (int ctp = 0; ctp < 2; ++ctp) {
            const int ct0 = ctp * 2, ct1 = ctp * 2 + 1;
            v8f acc0 = {}, acc1 = {};
#pragma unroll
            for (int kc = 0; kc < 4; ++kc) {
                // B 32x16 in the same contiguous k order as A: lane holds col (lane%16); element j -> K =
                // kc*32 + 16*(lane>=16) + j  => 16 contiguous bf16 per lane.
                const __bf16* ph0 = &sBh[ct0 * 16 + l15][kc * 32 + lh * 16];
                const __bf16* pl0 = &sBl[ct0 * 16 + l15][kc * 32 + lh * 16];
                const __bf16* ph1 = &sBh[ct1 * 16 + l15][kc * 32 + lh * 16];
                const __bf16* pl1 = &sBl[ct1 * 16 + l15][kc * 32 + lh * 16];
                union { v16bf v; v8bf h[2]; } bh0, bl0, bh1, bl1;
                bh0.h[0] = *(const v8bf*)ph0;  bh0.h[1] = *(const v8bf*)(ph0 + 8);
                bl0.h[0] = *(const v8bf*)pl0;  bl0.h[1] = *(const v8bf*)(pl0 + 8);
                bh1.h[0] = *(const v8bf*)ph1;  bh1.h[1] = *(const v8bf*)(ph1 + 8);
                bl1.h[0] = *(const v8bf*)pl1;  bl1.h[1] = *(const v8bf*)(pl1 + 8);
                acc0 = __builtin_amdgcn_wmma_f32_16x16x32_bf16(
                           false, ah[kc], false, bh0.v, (short)0, acc0, false, false);
                acc1 = __builtin_amdgcn_wmma_f32_16x16x32_bf16(
                           false, ah[kc], false, bh1.v, (short)0, acc1, false, false);
                acc0 = __builtin_amdgcn_wmma_f32_16x16x32_bf16(
                           false, al[kc], false, bh0.v, (short)0, acc0, false, false);
                acc1 = __builtin_amdgcn_wmma_f32_16x16x32_bf16(
                           false, al[kc], false, bh1.v, (short)0, acc1, false, false);
                acc0 = __builtin_amdgcn_wmma_f32_16x16x32_bf16(
                           false, ah[kc], false, bl0.v, (short)0, acc0, false, false);
                acc1 = __builtin_amdgcn_wmma_f32_16x16x32_bf16(
                           false, ah[kc], false, bl1.v, (short)0, acc1, false, false);
            }
            // scores (base-2): s = tcoef[m] + (2/(eps*ln2)) * dot
            const float tv0 = sT[ct0 * 16 + l15];
            const float tv1 = sT[ct1 * 16 + l15];
#pragma unroll
            for (int i = 0; i < 8; ++i) {
                float s0 = fmaf(kScale2, acc0[i], tv0);
                float s1 = fmaf(kScale2, acc1[i], tv1);
                float sb = fmaxf(s0, s1);
                float mo = rmax[i];
                float mn = fmaxf(mo, sb);
                rsum[i] = rsum[i] * __builtin_amdgcn_exp2f(mo - mn)
                          + __builtin_amdgcn_exp2f(s0 - mn)
                          + __builtin_amdgcn_exp2f(s1 - mn);
                rmax[i] = mn;
            }
        }
    }

    // cross-lane combine within each 16-lane half (row lives on 16 lanes)
#pragma unroll
    for (int i = 0; i < 8; ++i) {
        float gm = rmax[i];
#pragma unroll
        for (int off = 1; off < 16; off <<= 1)
            gm = fmaxf(gm, __shfl_xor(gm, off, 32));
        float c = rsum[i] * __builtin_amdgcn_exp2f(rmax[i] - gm);
#pragma unroll
        for (int off = 1; off < 16; off <<= 1)
            c += __shfl_xor(c, off, 32);
        if (l15 == 0) {
            int n = rowbase + i + lh * 8;
            pmax[(size_t)n * kMS + blockIdx.y] = gm;
            psum[(size_t)n * kMS + blockIdx.y] = c;
        }
    }
}

// ---- combine M-split partials into the final output ------------------------
__global__ __launch_bounds__(256) void ot_combine(const float* __restrict__ pmax,
                                                  const float* __restrict__ psum,
                                                  const float* __restrict__ xsq,
                                                  const float* __restrict__ meanp,
                                                  float* __restrict__ out) {
    int n = blockIdx.x * blockDim.x + threadIdx.x;
    const float* pm = pmax + (size_t)n * kMS;
    const float* ps = psum + (size_t)n * kMS;
    float gm = pm[0];
#pragma unroll
    for (int j = 1; j < kMS; ++j) gm = fmaxf(gm, pm[j]);
    float gs = 0.f;
#pragma unroll
    for (int j = 0; j < kMS; ++j)
        gs += ps[j] * __builtin_amdgcn_exp2f(pm[j] - gm);
    // -eps * ln(sum_m e^u) = -eps*ln2*(gm + log2(gs)); add back x^2 and mean(psi)
    out[n] = kNegEln2 * (gm + __builtin_amdgcn_logf(gs)) + xsq[n] + meanp[0];
}

extern "C" void kernel_launch(void* const* d_in, const int* in_sizes, int n_in,
                              void* d_out, int out_size, void* d_ws, size_t ws_size,
                              hipStream_t stream) {
    (void)in_sizes; (void)n_in; (void)out_size; (void)ws_size;
    const float* src = (const float*)d_in[0];   // [8192 x 128]
    const float* tgt = (const float*)d_in[1];   // [16384 x 128]
    const float* psi = (const float*)d_in[2];   // [16384]
    float* out = (float*)d_out;                 // [8192]

    // workspace layout (floats): xsq[8192] | tcoef[16384] | pmax[8192*8] |
    // psum[8192*8] | mean[1]   (~620 KB total)
    float* xsq   = (float*)d_ws;
    float* tco   = xsq + kN;
    float* pmax  = tco + kM;
    float* psum  = pmax + (size_t)kN * kMS;
    float* meanp = psum + (size_t)kN * kMS;

    ot_prep_xsq  <<<kN / 256, 256, 0, stream>>>(src, xsq);
    ot_prep_tcoef<<<kM / 256, 256, 0, stream>>>(tgt, psi, tco);
    ot_prep_mean <<<1, 256, 0, stream>>>(psi, meanp);

    dim3 grid(kN / kRowsWG, kMS);               // 64 x 8 workgroups, 8 waves each
    ot_main<<<grid, kTPB, 0, stream>>>(src, tgt, tco, pmax, psum);

    ot_combine<<<kN / 256, 256, 0, stream>>>(pmax, psum, xsq, meanp, out);
}
